// LS2T_41283225649480
// MI455X (gfx1250) — matrix operand order repaired and checked by reading, not compiled, not run on hardware
//
#include <hip/hip_runtime.h>

typedef __attribute__((ext_vector_type(16))) _Float16 v16h;
typedef __attribute__((ext_vector_type(8)))  _Float16 v8h;
typedef __attribute__((ext_vector_type(8)))  float    v8f;
typedef __attribute__((ext_vector_type(4)))  float    v4f;
typedef __attribute__((ext_vector_type(2)))  float    v2f;

#define B_DIM   64
#define T_LEN   4096
#define D_DIM   128
#define F_DIM   64
#define NCOMP   6
#define CHUNK   256
#define NCHUNK  (T_LEN / CHUNK)     // 16
#define SUBT    16                  // t rows per WMMA tile (M)
#define NITER   (CHUNK / (2*SUBT))  // 8 iterations of 32 rows
#define KT_STRIDE 136               // f16 per (c,f) row: 128 + 8 pad (bank spread)
#define KT_ELEMS  (NCOMP * F_DIM * KT_STRIDE)   // 52224 f16 = 104448 B
#define STAGE_F   (SUBT * 96)       // floats per wave: 16 t * (3 c-pairs * 16 f * 2)
#define NQ 10                       // per-chunk reduction quantities per f

// Fused chunk kernel: per block = one (batch b, time-chunk j).
// 4 waves; wave wf owns f-columns [wf*16, wf*16+16) for all 6 components.
__global__ void __launch_bounds__(128)
ls2t_chunk_kernel(const float* __restrict__ seq,
                  const float* __restrict__ kern,
                  const float* __restrict__ bias,
                  float* __restrict__ ws)
{
    extern __shared__ char smem_raw[];
    _Float16* kt      = (_Float16*)smem_raw;
    float*    stg_all = (float*)(smem_raw + KT_ELEMS * sizeof(_Float16));

    const int tid = threadIdx.x;
    const int wf  = tid >> 5;       // wave id 0..3
    const int l   = tid & 31;       // lane
    const int n   = l & 15;         // N column / t-row within halves
    const int kh  = l >> 4;         // k-half select
    const int j   = blockIdx.x;     // chunk index
    const int b   = blockIdx.y;     // batch row

    float* stg = stg_all + wf * STAGE_F;   // wave-private staging

    // ---- transpose + f32->f16 convert kernel matrix into LDS: kt[c][f][k] ----
    for (int i = tid; i < NCOMP * D_DIM * F_DIM; i += 128) {
        int c = i >> 13;            // / (128*64)
        int r = i & 8191;
        int d = r >> 6;
        int f = r & 63;
        kt[(c * F_DIM + f) * KT_STRIDE + d] = (_Float16)kern[i];
    }
    __syncthreads();                // only cross-wave sync in the kernel

    const int f = wf * 16 + n;      // lane's f column (mirrored across kh)

    float bv[NCOMP];
#pragma unroll
    for (int c = 0; c < NCOMP; ++c) bv[c] = bias[c * F_DIM + f];

    // chunk-local scan state (valid in lanes 0..15; lanes 16..31 mirror)
    float y0 = 0.f, y1 = 0.f, y2 = 0.f;      // local iterated sums
    float c1 = 0.f, s1 = 0.f, s2 = 0.f;      // exclusive prefix states
    float c4 = 0.f;                          // excl cumsum of M4 (end == SUM M4)
    float a2 = 0.f, a5 = 0.f, b45 = 0.f;     // SUM M2, SUM M5, SUM M5*cum(M4)

    const size_t seq_base = ((size_t)b * T_LEN + (size_t)j * CHUNK) * D_DIM;

    // Build one 16x32 f16 A fragment (lane%16 = M row, lane/16 = k-half)
    auto buildA = [&](const float* ap) {
        v4f x0 = *(const v4f*)(ap);
        v4f x1 = *(const v4f*)(ap + 4);
        v4f x2 = *(const v4f*)(ap + 16);
        v4f x3 = *(const v4f*)(ap + 20);
        v16h A;
#pragma unroll
        for (int e = 0; e < 4; ++e) {
            A[e]      = (_Float16)x0[e];
            A[4 + e]  = (_Float16)x1[e];
            A[8 + e]  = (_Float16)x2[e];
            A[12 + e] = (_Float16)x3[e];
        }
        return A;
    };

    // Stage one 16-row C-tile set to wave-private LDS and scan it sequentially.
    // Staging layout [t][c-pair][f][2] -> scan reads 3x ds_load_b64, bank-free.
    auto stageScan = [&](v8f* acc) {
        asm volatile("" ::: "memory");
#pragma unroll
        for (int c = 0; c < NCOMP; ++c)
#pragma unroll
            for (int r = 0; r < 8; ++r)
                stg[(r + 8 * kh) * 96 + (c >> 1) * 32 + n * 2 + (c & 1)] = acc[c][r];
        asm volatile("" ::: "memory");
        __builtin_amdgcn_wave_barrier();   // DS pipe is in-order per wave
        for (int tl = 0; tl < SUBT; ++tl) {
            const float* mrow = stg + tl * 96 + n * 2;
            v2f m01 = *(const v2f*)(mrow);
            v2f m23 = *(const v2f*)(mrow + 32);
            v2f m45 = *(const v2f*)(mrow + 64);
            float m0 = m01[0], m1 = m01[1];
            float m2 = m23[0], m3 = m23[1];
            float m4 = m45[0], m5 = m45[1];
            y2  = fmaf(m5, s2, y2);     // uses exclusive s2
            b45 = fmaf(m5, c4, b45);    // uses exclusive cum(M4)
            a5 += m5;
            s2  = fmaf(m4, s1, s2);     // uses exclusive s1
            c4 += m4;
            s1 += m3;
            y1  = fmaf(m2, c1, y1);     // uses exclusive c1
            a2 += m2;
            c1 += m1;
            y0 += m0;
        }
        asm volatile("" ::: "memory");
        __builtin_amdgcn_wave_barrier();
    };

    for (int it = 0; it < NITER; ++it) {     // 8 x 32 time rows
        v8f acc0[NCOMP], acc1[NCOMP];
#pragma unroll
        for (int c = 0; c < NCOMP; ++c)
#pragma unroll
            for (int e = 0; e < 8; ++e) { acc0[c][e] = bv[c]; acc1[c][e] = bv[c]; }

        const int    trow  = it * 2 * SUBT + n;
        const float* arow0 = seq + seq_base + (size_t)trow * D_DIM;
        const float* arow1 = arow0 + (size_t)SUBT * D_DIM;

        if (it + 1 < NITER)   // speculative prefetch of next iteration's rows
            __builtin_prefetch(arow0 + (size_t)(2 * SUBT) * D_DIM, 0, 1);

#pragma unroll
        for (int kk = 0; kk < 4; ++kk) {     // K = 128 in 4 steps of 32
            v16h A0 = buildA(arow0 + kk * 32 + kh * 8);
            v16h A1 = buildA(arow1 + kk * 32 + kh * 8);
#pragma unroll
            for (int c = 0; c < NCOMP; ++c) {
                // B layout: lane%16 = N col, lane/16 = k-half, K = 8*kh+0..7 then 16+8*kh+0..7
                const _Float16* bp = &kt[(c * F_DIM + f) * KT_STRIDE + kk * 32 + kh * 8];
                v8h b0 = *(const v8h*)bp;
                v8h b1 = *(const v8h*)(bp + 16);
                v16h Bv;
#pragma unroll
                for (int e = 0; e < 8; ++e) { Bv[e] = b0[e]; Bv[8 + e] = b1[e]; }
                acc0[c] = __builtin_amdgcn_wmma_f32_16x16x32_f16(
                    false, A0, false, Bv, (short)0, acc0[c], false, false);
                acc1[c] = __builtin_amdgcn_wmma_f32_16x16x32_f16(
                    false, A1, false, Bv, (short)0, acc1[c], false, false);
            }
        }

        stageScan(acc0);    // rows it*32 .. it*32+15
        stageScan(acc1);    // rows it*32+16 .. it*32+31
    }

    // per-chunk reductions -> workspace, coalesced over f
    if (l < 16) {
        float* wsp = ws + (size_t)(b * NCHUNK + j) * NQ * F_DIM + f;
        wsp[0 * F_DIM] = y0;
        wsp[1 * F_DIM] = c1;    // SUM M1
        wsp[2 * F_DIM] = a2;    // SUM M2
        wsp[3 * F_DIM] = s1;    // SUM M3
        wsp[4 * F_DIM] = c4;    // SUM M4
        wsp[5 * F_DIM] = a5;    // SUM M5
        wsp[6 * F_DIM] = y1;
        wsp[7 * F_DIM] = s2;    // local s2 end value
        wsp[8 * F_DIM] = b45;   // SUM M5*cum(M4)
        wsp[9 * F_DIM] = y2;
    }
}

// Sequential composition of the 16 chunk-linear-maps per (b,f) chain.
__global__ void __launch_bounds__(128)
ls2t_combine_kernel(const float* __restrict__ ws, float* __restrict__ out)
{
    int g = blockIdx.x * blockDim.x + threadIdx.x;
    if (g >= B_DIM * F_DIM) return;
    int b = g / F_DIM;
    int f = g % F_DIM;

    float Y0 = 0.f, Y1 = 0.f, Y2 = 0.f;
    float C1 = 0.f, S1 = 0.f, S2 = 0.f;
    for (int j = 0; j < NCHUNK; ++j) {
        const float* p = ws + (size_t)(b * NCHUNK + j) * NQ * F_DIM + f;
        float y0  = p[0 * F_DIM], c1 = p[1 * F_DIM], a2 = p[2 * F_DIM];
        float s1  = p[3 * F_DIM], a4 = p[4 * F_DIM], a5 = p[5 * F_DIM];
        float y1  = p[6 * F_DIM], s2 = p[7 * F_DIM], b45 = p[8 * F_DIM];
        float y2  = p[9 * F_DIM];
        Y0 += y0;
        Y1 += y1 + C1 * a2;
        Y2 += y2 + S2 * a5 + S1 * b45;
        S2 += s2 + S1 * a4;      // carry uses pre-update S1
        S1 += s1;
        C1 += c1;
    }
    float* o = out + (size_t)g * 3;   // [B][F][ORDER]
    o[0] = Y0; o[1] = Y1; o[2] = Y2;
}

extern "C" void kernel_launch(void* const* d_in, const int* in_sizes, int n_in,
                              void* d_out, int out_size, void* d_ws, size_t ws_size,
                              hipStream_t stream) {
    const float* seq  = (const float*)d_in[0];   // [64][4096][128] f32
    const float* kern = (const float*)d_in[1];   // [6][128][64]    f32
    const float* bias = (const float*)d_in[2];   // [6][64]         f32
    float* out = (float*)d_out;                  // [64][64][3]     f32
    float* ws  = (float*)d_ws;                   // 64*16*10*64 f32 = 2.62 MB

    const size_t lds_bytes = KT_ELEMS * sizeof(_Float16) + 4 * STAGE_F * sizeof(float);
    dim3 grid1(NCHUNK, B_DIM);
    ls2t_chunk_kernel<<<grid1, dim3(128), lds_bytes, stream>>>(seq, kern, bias, ws);

    int threads = B_DIM * F_DIM;
    ls2t_combine_kernel<<<dim3((threads + 127) / 128), dim3(128), 0, stream>>>(ws, out);
}
